// KITLSTM_M_24189255811657
// MI455X (gfx1250) — matrix operand order repaired and checked by reading, not compiled, not run on hardware
//
#include <hip/hip_runtime.h>
#include <stdint.h>

// ---------------- problem constants ----------------
#define BSZ 256
#define TSZ 512
#define ISZ 128
#define HSZ 256
#define SSZ 32
#define ONTN 20

// ---------------- GEMM constants -------------------
// Combined weight matrix: 8 row-blocks of 256 rows:
//   q=0: i  (W_ii | W_hi)      q=4: kg (W_ic | W_hc | W_pc)
//   q=1: f  (W_if | W_hf)      q=5: gd (W_g on u-segment)
//   q=2: g  (W_ig | W_hg)      q=6: cs (W_d on c-segment)
//   q=3: o  (W_io | W_ho)      q=7: k  (W_k on a-segment)
// K segments: x[0,128) u[128,256) h[256,512) c[512,768) p[768,800) a[800,832)
#define KTOT 832
#define KHALF 416
#define MROWS 2048
#define NBLOCKS 64   // 16 M-blocks(128 rows) x 4 N-blocks(64 cols)

typedef _Float16 f16;
typedef __attribute__((ext_vector_type(16))) _Float16 v16h;
typedef __attribute__((ext_vector_type(8)))  _Float16 v8h;
typedef __attribute__((ext_vector_type(8)))  float    v8f;

__device__ __forceinline__ float sigmoidf(float x){ return 1.0f/(1.0f + __expf(-x)); }

// ---------------- device-wide barrier ----------------
__device__ __forceinline__ void grid_barrier(unsigned* bar){
  __syncthreads();
  if (threadIdx.x == 0){
    __threadfence();
    volatile unsigned* gen = bar + 1;
    unsigned g = *gen;
    if (atomicAdd(bar, 1u) == (unsigned)(NBLOCKS - 1)){
      atomicExch(bar, 0u);
      __threadfence();
      atomicAdd((unsigned*)(bar + 1), 1u);
    } else {
      while (*gen == g) { __builtin_amdgcn_s_sleep(2); }
    }
    __threadfence();
  }
  __syncthreads();
}

// ---------------- phase-1 kernels ----------------
__global__ void kit_prep_data(const float* __restrict__ x, const float* __restrict__ dx,
                              f16* __restrict__ Xh, f16* __restrict__ Uh, int n){
  int i = blockIdx.x * blockDim.x + threadIdx.x;
  if (i < n){
    Xh[i] = (f16)x[i];
    Uh[i] = (f16)(1.0f / sigmoidf(dx[i]));
  }
}

// a_t = softmax_over_ONT(ent * p_t); written directly to output region (ONT,TS,BS)
__global__ void kit_prep_a(const float* __restrict__ binx, const float* __restrict__ emb,
                           float* __restrict__ aout){
  int idx = blockIdx.x * blockDim.x + threadIdx.x;   // BS*TS = 131072
  int b = idx >> 9;
  int t = idx & 511;
  float beta[ONTN];
  float m = -1e30f;
  #pragma unroll
  for (int j = 0; j < ONTN; ++j){
    beta[j] = emb[j] * binx[((size_t)b * TSZ + t) * ONTN + j];
    m = fmaxf(m, beta[j]);
  }
  float s = 0.f;
  #pragma unroll
  for (int j = 0; j < ONTN; ++j){ beta[j] = __expf(beta[j] - m); s += beta[j]; }
  float inv = 1.0f / s;
  #pragma unroll
  for (int j = 0; j < ONTN; ++j)
    aout[((size_t)j * TSZ + t) * BSZ + b] = beta[j] * inv;
}

// assemble block-structured Wbig (f16) + fused bias vector (fp32)
__global__ void kit_build_w(
    const float* Wii, const float* Wif, const float* Wig, const float* Wio,
    const float* Whi, const float* Whf, const float* Whg, const float* Who,
    const float* Wic, const float* Whc, const float* Wpc, const float* Wg,
    const float* Wd,  const float* Wk,
    const float* bii, const float* bif, const float* big_, const float* bio,
    const float* bhi, const float* bhf, const float* bhg, const float* bho,
    const float* bic, const float* bhc, const float* bpc, const float* bg,
    const float* bd,  const float* bk,
    f16* __restrict__ Wbig, float* __restrict__ bias)
{
  int row = blockIdx.x;        // 0..2047
  int q = row >> 8;
  int r = row & 255;
  const float* WX = nullptr; const float* WH = nullptr;
  switch (q){
    case 0: WX = Wii; WH = Whi; break;
    case 1: WX = Wif; WH = Whf; break;
    case 2: WX = Wig; WH = Whg; break;
    case 3: WX = Wio; WH = Who; break;
    case 4: WX = Wic; WH = Whc; break;
    default: break;
  }
  for (int k = threadIdx.x; k < KTOT; k += blockDim.x){
    float w = 0.f;
    if (k < 128)      { if (WX) w = WX[r * ISZ + k]; }
    else if (k < 256) { if (q == 5) w = Wg[r * ISZ + (k - 128)]; }
    else if (k < 512) { if (WH) w = WH[r * HSZ + (k - 256)]; }
    else if (k < 768) { if (q == 6) w = Wd[r * HSZ + (k - 512)]; }
    else if (k < 800) { int j = k - 768; if (q == 4 && j < ONTN) w = Wpc[r * ONTN + j]; }
    else              { int j = k - 800; if (q == 7 && j < ONTN) w = Wk[r * ONTN + j]; }
    Wbig[(size_t)row * KTOT + k] = (f16)w;
  }
  if (threadIdx.x == 0){
    float bb = 0.f;
    switch (q){
      case 0: bb = bii[r] + bhi[r]; break;
      case 1: bb = bif[r] + bhf[r]; break;
      case 2: bb = big_[r] + bhg[r]; break;
      case 3: bb = bio[r] + bho[r]; break;
      case 4: bb = bic[r] + bhc[r] + bpc[r]; break;
      case 5: bb = bg[r]; break;
      case 6: bb = bd[r]; break;
      case 7: bb = bk[r]; break;
    }
    bias[row] = bb;
  }
}

// initialize recurrent state (h0=c0=1, h_sum=0) and barrier counters
__global__ void kit_init(f16* hbT, f16* cbT, float* c32, float* hsum, unsigned* bar){
  int i = blockIdx.x * blockDim.x + threadIdx.x;
  if (i < HSZ * BSZ){
    hbT[i]  = (f16)1.0f;
    cbT[i]  = (f16)1.0f;
    c32[i] = 1.0f;
    hsum[i] = 0.0f;
  }
  if (i < 8) bar[i] = 0u;
}

// ---------------- persistent recurrence kernel ----------------
// hbT/cbT hold the f16 state mirrors in TRANSPOSED layout [b][r] so the
// B-panel staging (K-contiguous per batch column) vectorizes to b128 copies.
__global__ __launch_bounds__(256, 1) void kit_recur(
    const f16* __restrict__ Wbig, const f16* __restrict__ Xh, const f16* __restrict__ Uh,
    const float* __restrict__ binx, const float* __restrict__ aall,
    const float* __restrict__ d3,   const float* __restrict__ bias,
    f16* __restrict__ hbT, f16* __restrict__ cbT,
    float* __restrict__ c32, float* __restrict__ hsum,
    float* __restrict__ gout, unsigned* bar)
{
  __shared__ __align__(16) f16 Bsh[KHALF * 64];   // 52 KB staged B panel (half K)

  const int tid  = threadIdx.x;
  const int lane = tid & 31;
  const int wave = tid >> 5;
  const int mblk = blockIdx.x & 15;   // 16 M-blocks of 128 rows
  const int nblk = blockIdx.x >> 4;   // 4  N-blocks of 64 cols
  const int nbase = nblk * 64;

  // A fragment addressing (contiguous k order chosen here, B uses the same; not AMD's table order):
  //   lanes 0-15: row M=lane,    K = 0..15
  //   lanes 16-31: row M=lane-16, K = 16..31
  const int mrow  = mblk * 128 + wave * 16 + (lane & 15);
  const int koffA = (lane & 16) ? 16 : 0;
  // B fragment addressing (same contiguous k order as A): lanes 0-15 K=0..15, lanes 16-31 K=16..31
  const int khB   = (lane & 16) ? 16 : 0;

  // staging split: 4 threads per column, 13 chunks of 8 halves (16B) each
  const int scol = tid >> 2;           // column 0..63
  const int skl0 = (tid & 3) * 104;    // k-range start within half
  const int sb   = nbase + scol;       // batch column
  const size_t xrow = (size_t)sb * (TSZ * ISZ);
  const size_t hrow = (size_t)sb * HSZ;   // row base in transposed state mirrors

  for (int t = 0; t < TSZ; ++t){
    v8f acc0 = {}, acc1 = {}, acc2 = {}, acc3 = {};
    for (int half = 0; half < 2; ++half){
      // ---- stage B panel half into LDS (16-byte vector copies) ----
      {
        const size_t xbase = xrow + (size_t)t * ISZ;
        #pragma unroll 1
        for (int jj = 0; jj < 13; ++jj){
          int kl0 = skl0 + jj * 8;           // multiple of 8, chunk never straddles a segment
          int k0  = half * KHALF + kl0;
          const f16* src = nullptr;
          if (k0 < 128)       src = Xh + xbase + k0;
          else if (k0 < 256)  src = Uh + xbase + (k0 - 128);
          else if (k0 < 512)  src = hbT + hrow + (k0 - 256);
          else if (k0 < 768)  src = cbT + hrow + (k0 - 512);
          v8h v;
          if (src){
            v = *(const v8h*)src;
          } else if (k0 < 800){              // p segment (binary_x, pad to 32)
            #pragma unroll
            for (int u = 0; u < 8; ++u){
              int j = k0 - 768 + u;
              v[u] = (j < ONTN) ? (f16)binx[((size_t)sb * TSZ + t) * ONTN + j] : (f16)0.f;
            }
          } else {                           // a segment (softmax output, pad to 32)
            #pragma unroll
            for (int u = 0; u < 8; ++u){
              int j = k0 - 800 + u;
              v[u] = (j < ONTN) ? (f16)aall[((size_t)j * TSZ + t) * BSZ + sb] : (f16)0.f;
            }
          }
          *(v8h*)&Bsh[scol * KHALF + kl0] = v;
        }
      }
      __syncthreads();

      // prefetch next timestep's x panel into cache while we compute
      if (half == 0 && (t + 1) < TSZ)
        __builtin_prefetch((const void*)(Xh + xrow + (size_t)(t + 1) * ISZ), 0, 1);

      // ---- WMMA over 13 K-tiles of 32 ----
      // #pragma unroll 1 keeps one live accumulator set (C/D register-tied, no mov churn)
      const f16* arow = Wbig + (size_t)mrow * KTOT + half * KHALF + koffA;
      const f16* bbase = &Bsh[(lane & 15) * KHALF + khB];
      #pragma unroll 1
      for (int kk = 0; kk < 13; ++kk){
        v8h a0 = *(const v8h*)(arow + kk * 32);
        v8h a1 = *(const v8h*)(arow + kk * 32 + 8);
        v16h A;
        #pragma unroll
        for (int j = 0; j < 8; ++j){ A[j] = a0[j]; A[8 + j] = a1[j]; }

        const f16* bcol0 = bbase + kk * 32;
        v8h b00 = *(const v8h*)(bcol0);
        v8h b01 = *(const v8h*)(bcol0 + 8);
        v8h b10 = *(const v8h*)(bcol0 + 16 * KHALF);
        v8h b11 = *(const v8h*)(bcol0 + 16 * KHALF + 8);
        v8h b20 = *(const v8h*)(bcol0 + 32 * KHALF);
        v8h b21 = *(const v8h*)(bcol0 + 32 * KHALF + 8);
        v8h b30 = *(const v8h*)(bcol0 + 48 * KHALF);
        v8h b31 = *(const v8h*)(bcol0 + 48 * KHALF + 8);
        v16h B0, B1, B2, B3;
        #pragma unroll
        for (int j = 0; j < 8; ++j){
          B0[j] = b00[j]; B0[8 + j] = b01[j];
          B1[j] = b10[j]; B1[8 + j] = b11[j];
          B2[j] = b20[j]; B2[8 + j] = b21[j];
          B3[j] = b30[j]; B3[8 + j] = b31[j];
        }
        acc0 = __builtin_amdgcn_wmma_f32_16x16x32_f16(false, A, false, B0, (short)0, acc0, false, false);
        acc1 = __builtin_amdgcn_wmma_f32_16x16x32_f16(false, A, false, B1, (short)0, acc1, false, false);
        acc2 = __builtin_amdgcn_wmma_f32_16x16x32_f16(false, A, false, B2, (short)0, acc2, false, false);
        acc3 = __builtin_amdgcn_wmma_f32_16x16x32_f16(false, A, false, B3, (short)0, acc3, false, false);
      }
      __syncthreads();
    }

    // ---- store f32 result tiles (C/D layout: VGPR r -> M = r (+8 for hi lanes)) ----
    {
      const int rbase = mblk * 128 + wave * 16 + ((lane & 16) ? 8 : 0);
      const int cidx  = lane & 15;
      #pragma unroll
      for (int r = 0; r < 8; ++r){
        gout[(size_t)(rbase + r) * BSZ + nbase +  0 + cidx] = acc0[r];
        gout[(size_t)(rbase + r) * BSZ + nbase + 16 + cidx] = acc1[r];
        gout[(size_t)(rbase + r) * BSZ + nbase + 32 + cidx] = acc2[r];
        gout[(size_t)(rbase + r) * BSZ + nbase + 48 + cidx] = acc3[r];
      }
    }

    grid_barrier(bar);

    // ---- elementwise gate fusion + state update ----
    for (int e = blockIdx.x * 256 + tid; e < HSZ * BSZ; e += NBLOCKS * 256){
      int r = e >> 8;
      int b = e & 255;
      float vi  = sigmoidf(gout[e]              + bias[r]);
      float vf  = sigmoidf(gout[1 * 65536 + e]  + bias[256 + r]);
      float vg  = tanhf   (gout[2 * 65536 + e]  + bias[512 + r]);
      float vo  = sigmoidf(gout[3 * 65536 + e]  + bias[768 + r]);
      float vkg = tanhf   (gout[4 * 65536 + e]  + bias[1024 + r]);
      float vgd = sigmoidf(gout[5 * 65536 + e]  + bias[1280 + r]);
      float vcs = tanhf   (gout[6 * 65536 + e]  + bias[1536 + r]);
      float vk  = sigmoidf(gout[7 * 65536 + e]  + bias[1792 + r]);
      float g2  = sigmoidf(1.0f / d3[(size_t)b * TSZ + t]);
      float c     = c32[e];
      float cstar = (c - vcs) * vgd + vcs * g2;
      float cnew  = vf * cstar + vi * vg + vk * vkg;
      float hnew  = vo * tanhf(cnew);
      c32[e]   = cnew;
      hsum[e] += hnew;
      hbT[(size_t)b * HSZ + r] = (f16)hnew;   // transposed mirror for K-contiguous staging
      cbT[(size_t)b * HSZ + r] = (f16)cnew;
    }

    grid_barrier(bar);
  }
}

// ---------------- final head: W_y @ [h_sum; static] + batchnorm + sigmoid ----------------
__global__ void kit_final(const float* __restrict__ hsum, const float* __restrict__ statx,
                          const float* __restrict__ Wy, const float* __restrict__ by,
                          const float* __restrict__ emb,
                          const float* __restrict__ gamma, const float* __restrict__ beta,
                          float* __restrict__ out)
{
  __shared__ float red[256];
  int b = threadIdx.x;
  float s = by[0];
  for (int r = 0; r < HSZ; ++r) s += Wy[r] * hsum[r * BSZ + b];
  #pragma unroll
  for (int j = 0; j < SSZ; ++j) s += Wy[HSZ + j] * statx[b * SSZ + j];

  red[b] = s; __syncthreads();
  for (int st = 128; st > 0; st >>= 1){ if (b < st) red[b] += red[b + st]; __syncthreads(); }
  float mu = red[0] / 256.0f;
  __syncthreads();
  float d = s - mu;
  red[b] = d * d; __syncthreads();
  for (int st = 128; st > 0; st >>= 1){ if (b < st) red[b] += red[b + st]; __syncthreads(); }
  float var = red[0] / 256.0f;
  __syncthreads();

  out[b] = sigmoidf(d * rsqrtf(var + 1e-5f) * gamma[0] + beta[0]);

  if (b < ONTN){
    float e[ONTN]; float m = -1e30f;
    #pragma unroll
    for (int j = 0; j < ONTN; ++j){ e[j] = fabsf(emb[j] - emb[ONTN]); m = fmaxf(m, e[j]); }
    float sum = 0.f;
    #pragma unroll
    for (int j = 0; j < ONTN; ++j) sum += __expf(e[j] - m);
    out[BSZ + b] = __expf(e[b] - m) / sum;
  }
}

// ---------------- host-side orchestration ----------------
extern "C" void kernel_launch(void* const* d_in, const int* in_sizes, int n_in,
                              void* d_out, int out_size, void* d_ws, size_t ws_size,
                              hipStream_t stream) {
  (void)in_sizes; (void)n_in; (void)out_size; (void)ws_size;
  // inputs (setup_inputs order)
  const float* x     = (const float*)d_in[0];
  const float* dx    = (const float*)d_in[1];
  const float* binx  = (const float*)d_in[2];
  const float* statx = (const float*)d_in[3];
  const float* d3    = (const float*)d_in[4];
  // 5..8 unused (concept_dist, ont_*_emb)
  const float* Wii = (const float*)d_in[9];
  const float* Wif = (const float*)d_in[10];
  const float* Wig = (const float*)d_in[11];
  const float* Wio = (const float*)d_in[12];
  const float* Whi = (const float*)d_in[13];
  const float* Whf = (const float*)d_in[14];
  const float* Whg = (const float*)d_in[15];
  const float* Who = (const float*)d_in[16];
  const float* bii = (const float*)d_in[17];
  const float* bif = (const float*)d_in[18];
  const float* big_ = (const float*)d_in[19];
  const float* bio = (const float*)d_in[20];
  const float* bhi = (const float*)d_in[21];
  const float* bhf = (const float*)d_in[22];
  const float* bhg = (const float*)d_in[23];
  const float* bho = (const float*)d_in[24];
  const float* Wy  = (const float*)d_in[25];
  const float* by  = (const float*)d_in[26];
  const float* Wd  = (const float*)d_in[27];
  const float* bd  = (const float*)d_in[28];
  const float* Wg  = (const float*)d_in[29];
  const float* bg  = (const float*)d_in[30];
  const float* Wic = (const float*)d_in[31];
  const float* Whc = (const float*)d_in[32];
  const float* Wpc = (const float*)d_in[33];
  const float* bic = (const float*)d_in[34];
  const float* bhc = (const float*)d_in[35];
  const float* bpc = (const float*)d_in[36];
  const float* Wk  = (const float*)d_in[37];
  const float* bk  = (const float*)d_in[38];
  const float* emb = (const float*)d_in[39];
  const float* gamma = (const float*)d_in[40];
  const float* beta  = (const float*)d_in[41];

  float* out = (float*)d_out;                 // [y(256) | ont_dist_normed(20) | a(20,512,256)]
  float* aout = out + BSZ + ONTN;

  // workspace carve
  char* w = (char*)d_ws;
  auto carve = [&](size_t bytes) -> void* {
    void* p = (void*)w;
    w += (bytes + 255) & ~(size_t)255;
    return p;
  };
  const size_t NXE = (size_t)BSZ * TSZ * ISZ;   // 16,777,216
  f16*   Xh   = (f16*)  carve(NXE * 2);
  f16*   Uh   = (f16*)  carve(NXE * 2);
  f16*   Wbig = (f16*)  carve((size_t)MROWS * KTOT * 2);
  float* bias = (float*)carve((size_t)MROWS * 4);
  f16*   hbT  = (f16*)  carve((size_t)HSZ * BSZ * 2);
  f16*   cbT  = (f16*)  carve((size_t)HSZ * BSZ * 2);
  float* c32  = (float*)carve((size_t)HSZ * BSZ * 4);
  float* hsum = (float*)carve((size_t)HSZ * BSZ * 4);
  float* gout = (float*)carve((size_t)MROWS * BSZ * 4);
  unsigned* bar = (unsigned*)carve(256);

  // phase 1: parallel precompute
  kit_prep_data<<<(int)(NXE / 256), 256, 0, stream>>>(x, dx, Xh, Uh, (int)NXE);
  kit_prep_a<<<(BSZ * TSZ) / 256, 256, 0, stream>>>(binx, emb, aout);
  kit_build_w<<<MROWS, 256, 0, stream>>>(
      Wii, Wif, Wig, Wio, Whi, Whf, Whg, Who, Wic, Whc, Wpc, Wg, Wd, Wk,
      bii, bif, big_, bio, bhi, bhf, bhg, bho, bic, bhc, bpc, bg, bd, bk,
      Wbig, bias);
  kit_init<<<(HSZ * BSZ) / 256, 256, 0, stream>>>(hbT, cbT, c32, hsum, bar);

  // phase 2: persistent WMMA recurrence (512 steps, device-wide barriers)
  kit_recur<<<NBLOCKS, 256, 0, stream>>>(
      Wbig, Xh, Uh, binx, aout, d3, bias, hbT, cbT, c32, hsum, gout, bar);

  // phase 3: output head
  kit_final<<<1, 256, 0, stream>>>(hsum, statx, Wy, by, emb, gamma, beta, out);
}
